// gene_program_model_gcn_22651657519232
// MI455X (gfx1250) — matrix operand order repaired and checked by reading, not compiled, not run on hardware
//
#include <hip/hip_runtime.h>
#include <hip/hip_bf16.h>
#include <math.h>

// ---------------------------------------------------------------------------
// Types for CDNA5 WMMA (wave32): v_wmma_f32_16x16x32_bf16
// ---------------------------------------------------------------------------
typedef __bf16 bf16_t;
typedef __attribute__((ext_vector_type(16))) __bf16 v16bf;
typedef __attribute__((ext_vector_type(8)))  float  v8f;

#define DEVFN static __device__ __forceinline__

union FragAB { v16bf v; uint4 q[2]; };

// A-matrix 16x32 bf16 fragment (ISA 7.12.2): lane L -> row M = L&15,
// elements 0..7 at k = kb + 8*(L>>4) + e, elements 8..15 at +16.
// Row index is pre-clamped by the caller, so the load is unconditional.
DEVFN v16bf load_a_frag(const bf16_t* A, int row, int K, int kb, int lane) {
    FragAB f;
    const bf16_t* p = A + (size_t)row * K + kb + ((lane >> 4) << 3);
    f.q[0] = *(const uint4*)(p);        // 8 bf16 = 16B
    f.q[1] = *(const uint4*)(p + 16);   // second K half-block
    return f.v;
}

// B-matrix 32x16 bf16 fragment: lane L -> col N = L&15,
// elements 0..7 at k = kb + 8*(L>>4) + e, elements 8..15 at +16. B(k,n) = W[n,k] (row-major [Nout,K] weights),
// so each lane reads two 8-bf16 (16B) chunks from one weight row.
DEVFN v16bf load_b_frag(const bf16_t* W, int ncol, int K, int kb, int lane) {
    FragAB f;
    const bf16_t* p = W + (size_t)ncol * K + kb + ((lane >> 4) << 3);
    f.q[0] = *(const uint4*)(p);
    f.q[1] = *(const uint4*)(p + 16);
    return f.v;
}

// ---------------------------------------------------------------------------
// Fused WMMA GEMM:  out = act( A1 @ W1^T (+ A2 @ W2^T) + bias ) (+ resid)
// act: 0=none, 1=ELU, 2=softplus.
// One wave -> 16x64 output tile (4 accumulators reuse one A fragment).
// 8 waves/block -> 128 rows x 64 cols per block.
// ---------------------------------------------------------------------------
__global__ void __launch_bounds__(256) k_gemm_bf16(
    const bf16_t* __restrict__ A1, const bf16_t* __restrict__ W1, int K1,
    const bf16_t* __restrict__ A2, const bf16_t* __restrict__ W2, int K2,
    const float*  __restrict__ bias,
    const bf16_t* __restrict__ resid,
    bf16_t* __restrict__ outB, float* __restrict__ outF,
    int N, int Nout, int act)
{
    const int lane  = threadIdx.x & 31;
    const int wave  = threadIdx.x >> 5;
    const int tileM = blockIdx.x * 8 + wave;
    if (tileM * 16 >= N) return;                 // wave-uniform exit

    const int mLane = tileM * 16 + (lane & 15);  // A-fragment row for this lane
    // Clamp instead of zero-fill: row m of D depends only on row m of A, and
    // out-of-range rows are masked at the store, so clamping is safe and
    // keeps the inner loop free of EXEC-masked loads / zero fills.
    const int rowA  = (mLane < N) ? mLane : (N - 1);
    const int nBase = blockIdx.y * 64;
    const int nLane = lane & 15;

    v8f acc0 = {}, acc1 = {}, acc2 = {}, acc3 = {};

    for (int kb = 0; kb < K1; kb += 32) {
        v16bf a  = load_a_frag(A1, rowA, K1, kb, lane);
        v16bf b0 = load_b_frag(W1, nBase +  0 + nLane, K1, kb, lane);
        v16bf b1 = load_b_frag(W1, nBase + 16 + nLane, K1, kb, lane);
        v16bf b2 = load_b_frag(W1, nBase + 32 + nLane, K1, kb, lane);
        v16bf b3 = load_b_frag(W1, nBase + 48 + nLane, K1, kb, lane);
        acc0 = __builtin_amdgcn_wmma_f32_16x16x32_bf16(false, a, false, b0, (short)0, acc0, false, false);
        acc1 = __builtin_amdgcn_wmma_f32_16x16x32_bf16(false, a, false, b1, (short)0, acc1, false, false);
        acc2 = __builtin_amdgcn_wmma_f32_16x16x32_bf16(false, a, false, b2, (short)0, acc2, false, false);
        acc3 = __builtin_amdgcn_wmma_f32_16x16x32_bf16(false, a, false, b3, (short)0, acc3, false, false);
    }
    if (A2 != nullptr) {
        for (int kb = 0; kb < K2; kb += 32) {
            v16bf a  = load_a_frag(A2, rowA, K2, kb, lane);
            v16bf b0 = load_b_frag(W2, nBase +  0 + nLane, K2, kb, lane);
            v16bf b1 = load_b_frag(W2, nBase + 16 + nLane, K2, kb, lane);
            v16bf b2 = load_b_frag(W2, nBase + 32 + nLane, K2, kb, lane);
            v16bf b3 = load_b_frag(W2, nBase + 48 + nLane, K2, kb, lane);
            acc0 = __builtin_amdgcn_wmma_f32_16x16x32_bf16(false, a, false, b0, (short)0, acc0, false, false);
            acc1 = __builtin_amdgcn_wmma_f32_16x16x32_bf16(false, a, false, b1, (short)0, acc1, false, false);
            acc2 = __builtin_amdgcn_wmma_f32_16x16x32_bf16(false, a, false, b2, (short)0, acc2, false, false);
            acc3 = __builtin_amdgcn_wmma_f32_16x16x32_bf16(false, a, false, b3, (short)0, acc3, false, false);
        }
    }

    // Epilogue. C/D layout: VGPR v -> M = v + 8*(lane>=16), lane&15 -> N.
    const int hi = lane >> 4;
    const v8f* accs[4] = { &acc0, &acc1, &acc2, &acc3 };
#pragma unroll
    for (int t = 0; t < 4; ++t) {
        const int n  = nBase + t * 16 + nLane;
        const float bn = bias ? bias[n] : 0.0f;
        const v8f& acc = *accs[t];
#pragma unroll
        for (int v = 0; v < 8; ++v) {
            const int m = tileM * 16 + v + hi * 8;
            if (m >= N) continue;
            float xv = acc[v] + bn;
            if (act == 1) {                       // ELU (alpha = 1)
                xv = (xv > 0.0f) ? xv : (__expf(xv) - 1.0f);
            } else if (act == 2) {                // softplus, stable
                xv = (xv > 20.0f) ? xv : ((xv < -20.0f) ? __expf(xv) : log1pf(__expf(xv)));
            }
            if (resid) xv += (float)resid[(size_t)m * Nout + n];
            if (outB) outB[(size_t)m * Nout + n] = (bf16_t)xv;
            if (outF) outF[(size_t)m * Nout + n] = xv;
        }
    }
}

// ---------------------------------------------------------------------------
// Elementwise / graph kernels
// ---------------------------------------------------------------------------
__global__ void k_zero_f32(float* p, long long n) {
    long long i = blockIdx.x * (long long)blockDim.x + threadIdx.x;
    if (i < n) p[i] = 0.0f;
}

__global__ void k_cvt_f32_bf16(const float* __restrict__ s, bf16_t* __restrict__ d, long long n) {
    long long i = blockIdx.x * (long long)blockDim.x + threadIdx.x;
    if (i < n) d[i] = (bf16_t)s[i];
}

__global__ void k_degree(const int* __restrict__ dst, float* __restrict__ deg, long long E) {
    long long e = blockIdx.x * (long long)blockDim.x + threadIdx.x;
    if (e < E) atomicAdd(deg + dst[e], 1.0f);
}

// scatter-add f32 rows: agg[dst[e], :] += x[src[e], :]   (thread = edge x 4 feats)
__global__ void k_scatter_f32(const float* __restrict__ x,
                              const int* __restrict__ src, const int* __restrict__ dst,
                              float* __restrict__ agg, long long E, int d)
{
    long long i = blockIdx.x * (long long)blockDim.x + threadIdx.x;
    const int vpr = d >> 2;
    if (i >= E * (long long)vpr) return;
    long long e = i / vpr;
    int c = (int)(i - e * vpr) << 2;
    const float* xp = x + (size_t)src[e] * d + c;
    float* ap = agg + (size_t)dst[e] * d + c;
    atomicAdd(ap + 0, xp[0]);
    atomicAdd(ap + 1, xp[1]);
    atomicAdd(ap + 2, xp[2]);
    atomicAdd(ap + 3, xp[3]);
}

// scatter-add from bf16 source rows into f32 accumulator
__global__ void k_scatter_bf16(const bf16_t* __restrict__ x,
                               const int* __restrict__ src, const int* __restrict__ dst,
                               float* __restrict__ agg, long long E, int d)
{
    long long i = blockIdx.x * (long long)blockDim.x + threadIdx.x;
    const int vpr = d >> 2;
    if (i >= E * (long long)vpr) return;
    long long e = i / vpr;
    int c = (int)(i - e * vpr) << 2;
    const bf16_t* xp = x + (size_t)src[e] * d + c;
    float* ap = agg + (size_t)dst[e] * d + c;
    atomicAdd(ap + 0, (float)xp[0]);
    atomicAdd(ap + 1, (float)xp[1]);
    atomicAdd(ap + 2, (float)xp[2]);
    atomicAdd(ap + 3, (float)xp[3]);
}

// mean (divide by max(deg,1)) + convert to bf16
__global__ void k_finish_mean(const float* __restrict__ agg, const float* __restrict__ deg,
                              bf16_t* __restrict__ out, long long N, int d)
{
    long long i = blockIdx.x * (long long)blockDim.x + threadIdx.x;
    if (i >= N * (long long)d) return;
    long long row = i / d;
    float s = 1.0f / fmaxf(deg[row], 1.0f);
    out[i] = (bf16_t)(agg[i] * s);
}

// ---------------------------------------------------------------------------
// Host driver
// ---------------------------------------------------------------------------
static inline int divUp(long long a, int b) { return (int)((a + b - 1) / b); }

extern "C" void kernel_launch(void* const* d_in, const int* in_sizes, int n_in,
                              void* d_out, int out_size, void* d_ws, size_t ws_size,
                              hipStream_t stream)
{
    const int IN = 128, H = 256, OUT = 128;
    const long long N = in_sizes[0] / IN;
    const long long E = in_sizes[1] / 2;

    const float* x      = (const float*)d_in[0];
    const int*   ei     = (const int*)d_in[1];
    const int*   src    = ei;
    const int*   dst    = ei + E;
    const float* W_l1   = (const float*)d_in[2];
    const float* b_l1   = (const float*)d_in[3];
    const float* W_r1   = (const float*)d_in[4];
    const float* W_l2   = (const float*)d_in[5];
    const float* b_l2   = (const float*)d_in[6];
    const float* W_r2   = (const float*)d_in[7];
    const float* res1_W = (const float*)d_in[8];
    const float* res1_b = (const float*)d_in[9];
    const float* res2_W = (const float*)d_in[10];
    const float* res2_b = (const float*)d_in[11];
    const float* fc1_W  = (const float*)d_in[12];
    const float* fc1_b  = (const float*)d_in[13];
    const float* fc2_W  = (const float*)d_in[14];
    const float* fc2_b  = (const float*)d_in[15];
    const float* fc3_W  = (const float*)d_in[16];
    const float* fc3_b  = (const float*)d_in[17];
    float* out = (float*)d_out;

    // ---- workspace carve-up (256B aligned) ----
    char* wp = (char*)d_ws;
    auto alloc = [&](size_t bytes) -> void* {
        void* p = (void*)wp;
        wp += (bytes + 255) & ~(size_t)255;
        return p;
    };
    float*  deg   = (float*)alloc(N * 4);
    float*  agg   = (float*)alloc(N * (size_t)H * 4);     // f32 accumulator (reused)
    bf16_t* xb    = (bf16_t*)alloc(N * (size_t)IN * 2);
    bf16_t* h1b   = (bf16_t*)alloc(N * (size_t)H * 2);    // h1, later reused for h3
    bf16_t* aggb  = (bf16_t*)alloc(N * (size_t)H * 2);    // mean agg, later reused for h4
    bf16_t* res1b = (bf16_t*)alloc(N * (size_t)H * 2);
    bf16_t* res2b = (bf16_t*)alloc(N * (size_t)H * 2);
    bf16_t* h2b   = (bf16_t*)alloc(N * (size_t)H * 2);
    bf16_t* wl1b  = (bf16_t*)alloc((size_t)H * IN * 2);
    bf16_t* wr1b  = (bf16_t*)alloc((size_t)H * IN * 2);
    bf16_t* r1wb  = (bf16_t*)alloc((size_t)H * IN * 2);
    bf16_t* r2wb  = (bf16_t*)alloc((size_t)H * IN * 2);
    bf16_t* wl2b  = (bf16_t*)alloc((size_t)H * H * 2);
    bf16_t* wr2b  = (bf16_t*)alloc((size_t)H * H * 2);
    bf16_t* f1wb  = (bf16_t*)alloc((size_t)H * H * 2);
    bf16_t* f2wb  = (bf16_t*)alloc((size_t)H * H * 2);
    bf16_t* f3wb  = (bf16_t*)alloc((size_t)OUT * H * 2);

    const int T = 256;
    auto gr = [&](long long n) { return dim3((unsigned)divUp(n, T)); };

    // ---- precision conversion (once per launch) ----
    k_cvt_f32_bf16<<<gr(N * IN), T, 0, stream>>>(x, xb, N * IN);
    k_cvt_f32_bf16<<<gr((long long)H * IN), T, 0, stream>>>(W_l1, wl1b, (long long)H * IN);
    k_cvt_f32_bf16<<<gr((long long)H * IN), T, 0, stream>>>(W_r1, wr1b, (long long)H * IN);
    k_cvt_f32_bf16<<<gr((long long)H * IN), T, 0, stream>>>(res1_W, r1wb, (long long)H * IN);
    k_cvt_f32_bf16<<<gr((long long)H * IN), T, 0, stream>>>(res2_W, r2wb, (long long)H * IN);
    k_cvt_f32_bf16<<<gr((long long)H * H), T, 0, stream>>>(W_l2, wl2b, (long long)H * H);
    k_cvt_f32_bf16<<<gr((long long)H * H), T, 0, stream>>>(W_r2, wr2b, (long long)H * H);
    k_cvt_f32_bf16<<<gr((long long)H * H), T, 0, stream>>>(fc1_W, f1wb, (long long)H * H);
    k_cvt_f32_bf16<<<gr((long long)H * H), T, 0, stream>>>(fc2_W, f2wb, (long long)H * H);
    k_cvt_f32_bf16<<<gr((long long)OUT * H), T, 0, stream>>>(fc3_W, f3wb, (long long)OUT * H);

    // ---- degree (shared by both SAGE layers) ----
    k_zero_f32<<<gr(N), T, 0, stream>>>(deg, N);
    k_degree<<<gr(E), T, 0, stream>>>(dst, deg, E);

    dim3 blk(256);
    dim3 gH(divUp(N, 128), H / 64);    // 16x64 tile per wave
    dim3 gO(divUp(N, 128), OUT / 64);

    // ---- residual projections: res{1,2} = x @ resW^T + b ----
    k_gemm_bf16<<<gH, blk, 0, stream>>>(xb, r1wb, IN, nullptr, nullptr, 0,
                                        res1_b, nullptr, res1b, nullptr, (int)N, H, 0);
    k_gemm_bf16<<<gH, blk, 0, stream>>>(xb, r2wb, IN, nullptr, nullptr, 0,
                                        res2_b, nullptr, res2b, nullptr, (int)N, H, 0);

    // ---- SAGE layer 1: mean-aggregate x, then fused dual-GEMM + ELU ----
    k_zero_f32<<<gr(N * H), T, 0, stream>>>(agg, N * H);
    k_scatter_f32<<<gr(E * (IN / 4)), T, 0, stream>>>(x, src, dst, agg, E, IN);
    k_finish_mean<<<gr(N * IN), T, 0, stream>>>(agg, deg, aggb, N, IN);
    k_gemm_bf16<<<gH, blk, 0, stream>>>(aggb, wl1b, IN, xb, wr1b, IN,
                                        b_l1, nullptr, h1b, nullptr, (int)N, H, 1);

    // ---- SAGE layer 2: mean-aggregate h1, fused dual-GEMM + ELU + res1 ----
    k_zero_f32<<<gr(N * H), T, 0, stream>>>(agg, N * H);
    k_scatter_bf16<<<gr(E * (H / 4)), T, 0, stream>>>(h1b, src, dst, agg, E, H);
    k_finish_mean<<<gr(N * H), T, 0, stream>>>(agg, deg, aggb, N, H);
    k_gemm_bf16<<<gH, blk, 0, stream>>>(aggb, wl2b, H, h1b, wr2b, H,
                                        b_l2, res1b, h2b, nullptr, (int)N, H, 1);

    // ---- MLP head ----
    // h3 = elu(h2 @ fc1^T + b)         (h1b buffer reused for h3)
    k_gemm_bf16<<<gH, blk, 0, stream>>>(h2b, f1wb, H, nullptr, nullptr, 0,
                                        fc1_b, nullptr, h1b, nullptr, (int)N, H, 1);
    // h4 = elu(h3 @ fc2^T + b) + res2  (aggb buffer reused for h4)
    k_gemm_bf16<<<gH, blk, 0, stream>>>(h1b, f2wb, H, nullptr, nullptr, 0,
                                        fc2_b, res2b, aggb, nullptr, (int)N, H, 1);
    // out = softplus(h4 @ fc3^T + b)   -> f32 output
    k_gemm_bf16<<<gO, blk, 0, stream>>>(aggb, f3wb, H, nullptr, nullptr, 0,
                                        fc3_b, nullptr, nullptr, out, (int)N, OUT, 2);
}
